// ISTFT_83124797047170
// MI455X (gfx1250) — matrix operand order repaired and checked by reading, not compiled, not run on hardware
//
#include <hip/hip_runtime.h>
#include <hip/hip_bf16.h>

// ---------------- problem constants ----------------
#define NB     8          // batch
#define ND     513        // D = NFFT/2+1
#define NT     4096       // frames T
#define WINL   1024       // win_length
#define PADC   384        // (WIN-HOP)/2
#define K2D    1026       // 2*D
#define KPAD   1056       // K padded to multiple of 32 (33 k-tiles)
#define KTILES 33
#define LDSK   1080       // LDS row stride (bf16): 540 dwords = 28 mod 64 banks -> conflict-min
#define OUTN   1048576    // per-batch output length (L - 2*PAD)
#define MROWS  64         // t0 rows (output frames) per block
#define MTILES 4          // MROWS / 16
#define SROWS  67         // staged spec rows = MROWS + 3
#define TTILES 65         // ceil(4098/64): covers t0 0..4159 (need 1..4097)

typedef __bf16 v16bf __attribute__((ext_vector_type(16)));
typedef __bf16 v8bf  __attribute__((ext_vector_type(8)));
typedef float  v8f   __attribute__((ext_vector_type(8)));

typedef __attribute__((address_space(3))) const __bf16* lds_bf16p;
typedef __attribute__((address_space(3))) const v8bf*  lds_v8bfp;

static __device__ __forceinline__ v16bf combine16(v8bf lo, v8bf hi) {
    return __builtin_shufflevector(lo, hi, 0,1,2,3,4,5,6,7,8,9,10,11,12,13,14,15);
}

// ---------------- prep: kernel fp32 [K2D,WIN] -> bf16 transposed [WIN,KPAD] ----------------
__global__ __launch_bounds__(256)
void ISTFT_prep_kbT(const float* __restrict__ kern, __bf16* __restrict__ kbT) {
    int idx = blockIdx.x * 256 + threadIdx.x;
    if (idx >= WINL * KPAD) return;
    int n = idx / KPAD;
    int c = idx - n * KPAD;
    float v = (c < K2D) ? kern[(size_t)c * WINL + n] : 0.f;
    kbT[idx] = (__bf16)v;
}

// ---------------- main: spec build + 4x WMMA GEMM + overlap-add gather + normalize ----------------
__global__ __launch_bounds__(256)
void ISTFT_wmma(const float* __restrict__ mag, const float* __restrict__ ph,
                const __bf16* __restrict__ kbT, const float* __restrict__ wsq,
                float* __restrict__ out) {
    __shared__ __align__(16) __bf16 lspec[SROWS * LDSK];  // spec^T rows t0s-3 .. t0s+MROWS-1
    __shared__ float lwsq[WINL];

    const int tid = threadIdx.x;
    const int b   = blockIdx.y;
    const int t0s = blockIdx.x * MROWS;       // first output t0 row of this block
    const int t0b = t0s - 3;                  // t of LDS row 0

    // ---- stage spec (mag*cos, mag*sin) transposed into LDS, zero out-of-range t ----
    for (int e = tid; e < SROWS * ND; e += 256) {
        int row = e / ND;
        int c   = e - row * ND;
        int t   = t0b + row;
        float re = 0.f, im = 0.f;
        if ((unsigned)t < (unsigned)NT) {
            size_t gi = ((size_t)(b * ND + c)) * NT + t;
            float m = mag[gi];
            float p = ph[gi];
            float s, co;
            __sincosf(p, &s, &co);
            re = m * co;
            im = m * s;
        }
        lspec[row * LDSK + c]      = (__bf16)re;
        lspec[row * LDSK + ND + c] = (__bf16)im;
    }
    // zero-pad K region [K2D, LDSK)
    for (int e = tid; e < SROWS * (LDSK - K2D); e += 256) {
        int row = e / (LDSK - K2D);
        int c   = K2D + (e - row * (LDSK - K2D));
        lspec[row * LDSK + c] = (__bf16)0.f;
    }
    for (int i = tid; i < WINL; i += 256) lwsq[i] = wsq[i];
    __syncthreads();

    const int wv   = tid >> 5;     // wave 0..7 -> r columns [32*wv, 32*wv+32)
    const int lane = tid & 31;
    const int m_   = lane & 15;    // A row / B column within tile
    const int kh   = lane >> 4;    // K-half selector

    v8f acc[MTILES][2];            // [m-tile][n-tile] accumulators
    #pragma unroll
    for (int mt = 0; mt < MTILES; ++mt) {
        #pragma unroll
        for (int tt = 0; tt < 2; ++tt)
            acc[mt][tt] = (v8f){0.f,0.f,0.f,0.f,0.f,0.f,0.f,0.f};
    }

    #pragma unroll 1               // keep j rolled: compact loop-resident code
    for (int j = 0; j < 4; ++j) {
        const int rowoff = (m_ + 3 - j) * LDSK;
        // base for m-tiles 0,1 (disp 0 / 16*LDSK*2)
        lds_bf16p aA = (lds_bf16p)(lspec + rowoff);
        // base for m-tiles 2,3: opaque 32-bit LDS pointer the compiler cannot
        // re-fold into aA -> stays a dedicated base VGPR, loads use imm disp
        lds_bf16p aB = (lds_bf16p)(lspec + rowoff + 32 * LDSK);
        asm("" : "+v"(aB));
        const __bf16* bp = kbT + (size_t)(256 * j + 32 * wv + m_) * KPAD;
        #pragma unroll 3           // 33 = 3*11: no remainder, 24 WMMA per iteration
        for (int kt = 0; kt < KTILES; ++kt) {
            const int ka = kt * 32 + kh * 8;
            const int kb = kt * 32 + kh * 8;
            // load all fragments first -> clause-able loads, coalesced waits
            v16bf bb0 = combine16(*(const v8bf*)(bp + kb),
                                  *(const v8bf*)(bp + kb + 16));
            v16bf bb1 = combine16(*(const v8bf*)(bp + 16 * KPAD + kb),
                                  *(const v8bf*)(bp + 16 * KPAD + kb + 16));
            v16bf a0  = combine16(*(lds_v8bfp)(aA + ka),
                                  *(lds_v8bfp)(aA + ka + 16));
            v16bf a1  = combine16(*(lds_v8bfp)(aA + 16 * LDSK + ka),
                                  *(lds_v8bfp)(aA + 16 * LDSK + ka + 16));
            v16bf a2  = combine16(*(lds_v8bfp)(aB + ka),
                                  *(lds_v8bfp)(aB + ka + 16));
            v16bf a3  = combine16(*(lds_v8bfp)(aB + 16 * LDSK + ka),
                                  *(lds_v8bfp)(aB + 16 * LDSK + ka + 16));
            acc[0][0] = __builtin_amdgcn_wmma_f32_16x16x32_bf16(
                            false, a0, false, bb0, (short)0, acc[0][0], false, false);
            acc[0][1] = __builtin_amdgcn_wmma_f32_16x16x32_bf16(
                            false, a0, false, bb1, (short)0, acc[0][1], false, false);
            acc[1][0] = __builtin_amdgcn_wmma_f32_16x16x32_bf16(
                            false, a1, false, bb0, (short)0, acc[1][0], false, false);
            acc[1][1] = __builtin_amdgcn_wmma_f32_16x16x32_bf16(
                            false, a1, false, bb1, (short)0, acc[1][1], false, false);
            acc[2][0] = __builtin_amdgcn_wmma_f32_16x16x32_bf16(
                            false, a2, false, bb0, (short)0, acc[2][0], false, false);
            acc[2][1] = __builtin_amdgcn_wmma_f32_16x16x32_bf16(
                            false, a2, false, bb1, (short)0, acc[2][1], false, false);
            acc[3][0] = __builtin_amdgcn_wmma_f32_16x16x32_bf16(
                            false, a3, false, bb0, (short)0, acc[3][0], false, false);
            acc[3][1] = __builtin_amdgcn_wmma_f32_16x16x32_bf16(
                            false, a3, false, bb1, (short)0, acc[3][1], false, false);
        }
    }

    // ---- epilogue: normalize by window^2 OLA and store (each output written once) ----
    #pragma unroll
    for (int mt = 0; mt < MTILES; ++mt) {
        #pragma unroll
        for (int tt = 0; tt < 2; ++tt) {
            v8f a = acc[mt][tt];
            const int r = 32 * wv + 16 * tt + m_;        // s % 256
            #pragma unroll
            for (int v = 0; v < 8; ++v) {
                const int mm = 16 * mt + (kh << 3) + v;  // D-tile row within block
                const int t0 = t0s + mm;
                float coff = 0.f;
                #pragma unroll
                for (int j = 0; j < 4; ++j) {
                    int tj = t0 - j;
                    if ((unsigned)tj < (unsigned)NT) coff += lwsq[r + 256 * j];
                }
                long si = 256L * t0 + r - PADC;
                if (si >= 0 && si < (long)OUTN)
                    out[(size_t)b * OUTN + si] = a[v] / fmaxf(coff, 1e-8f);
            }
        }
    }
}

extern "C" void kernel_launch(void* const* d_in, const int* in_sizes, int n_in,
                              void* d_out, int out_size, void* d_ws, size_t ws_size,
                              hipStream_t stream) {
    (void)in_sizes; (void)n_in; (void)out_size; (void)ws_size;
    const float* mag  = (const float*)d_in[0];   // inputs    [B,D,T]
    const float* phs  = (const float*)d_in[1];   // phase     [B,D,T]
    const float* kern = (const float*)d_in[2];   // kernel    [2D,WIN]
    const float* wsq  = (const float*)d_in[3];   // window_sq [WIN]
    float* out = (float*)d_out;                  // [B,1,OUTN]
    __bf16* kbT = (__bf16*)d_ws;                 // [WIN,KPAD] bf16, 2.16 MB

    ISTFT_prep_kbT<<<(WINL * KPAD + 255) / 256, 256, 0, stream>>>(kern, kbT);

    dim3 grid(TTILES, NB);
    ISTFT_wmma<<<grid, 256, 0, stream>>>(mag, phs, kbT, wsq, out);
}
